// RAFT_40973988004127
// MI455X (gfx1250) — matrix operand order repaired and checked by reading, not compiled, not run on hardware
//
#include <hip/hip_runtime.h>
#include <hip/hip_bf16.h>
#include <math.h>

// Problem constants (match reference)
#define B_   2
#define C_   256
#define H8_  56
#define W8_  96
#define HW_  (H8_ * W8_)     // 5376
#define NQ_  (B_ * HW_)      // 10752

typedef __attribute__((ext_vector_type(16))) __bf16          v16bf;
typedef __attribute__((ext_vector_type(8)))  float           v8f;
typedef __attribute__((ext_vector_type(8)))  unsigned short  v8us;
typedef __attribute__((ext_vector_type(4)))  unsigned int    u32x4;
typedef __attribute__((ext_vector_type(8)))  int             i32x8;
typedef __attribute__((ext_vector_type(4)))  int             i32x4;

__device__ __forceinline__ unsigned short f32_to_bf16_rne(float f) {
    union { float f; unsigned int u; } v;
    v.f = f;
    unsigned int u = v.u;
    unsigned int r = u + 0x7FFFu + ((u >> 16) & 1u);
    return (unsigned short)(r >> 16);
}

// ---------------------------------------------------------------------------
// TDM: issue a 2D tile load (bf16 elements) into LDS via the Tensor Data
// Mover.  D# packing per CDNA5 ISA ch.8: group0 = {count/type/addr}, group1 =
// {data_size, tensor dims/strides, tile dims}.  2D tensor -> remaining
// descriptor groups zero.  One call per wave = one TENSOR_LOAD_TO_LDS
// instruction (EXEC ignored).  This toolchain exposes the 6-arg builtin:
// (u32x4 g0, i32x8 g1, i32x4, i32x4, i32x8, i32 cpol).
// ---------------------------------------------------------------------------
__device__ __forceinline__ void tdm_load_2d_bf16(
    unsigned lds_byte_off, const void* gaddr,
    unsigned tensor_d0, unsigned tensor_d1,
    unsigned tile_d0, unsigned tile_d1, unsigned stride_d0) {
    u32x4 g0;
    g0[0] = 1u;                                    // count=1, user mode
    g0[1] = lds_byte_off;                          // LDS byte address
    unsigned long long ga = (unsigned long long)(uintptr_t)gaddr;
    g0[2] = (unsigned)(ga & 0xFFFFFFFFu);          // global_addr[31:0]
    g0[3] = (unsigned)((ga >> 32) & 0x01FFFFFFu)   // global_addr[56:32]
          | (2u << 30);                            // type = 2 ("image")
    i32x8 g1;
    g1[0] = (int)(1u << 16);                       // data_size=1 (2 bytes)
    g1[1] = (int)((tensor_d0 & 0xFFFFu) << 16);    // tensor_dim0[15:0]
    g1[2] = (int)((tensor_d0 >> 16) | ((tensor_d1 & 0xFFFFu) << 16));
    g1[3] = (int)((tensor_d1 >> 16) | ((tile_d0 & 0xFFFFu) << 16));
    g1[4] = (int)(tile_d1 & 0xFFFFu);              // tile_dim1; tile_dim2=0
    g1[5] = (int)stride_d0;                        // tensor_dim0_stride[31:0]
    g1[6] = 0;
    g1[7] = 0;
    i32x4 z4 = {};
    i32x8 z8 = {};
    __builtin_amdgcn_tensor_load_to_lds(g0, g1, z4, z4, z8, 0);
}

// ---------------------------------------------------------------------------
// Kernel 0: convert f32 feature maps [B][C][HW] into bf16 transposed layout
// [B][HW][C] (k-contiguous rows) via LDS 32x32 tile transpose.
// ---------------------------------------------------------------------------
__global__ __launch_bounds__(256) void convert_transpose_bf16(
    const float* __restrict__ in, unsigned short* __restrict__ out) {
    __shared__ unsigned short tile[32][33];
    const int b  = blockIdx.z;
    const int m0 = blockIdx.x * 32;   // pixel tile
    const int c0 = blockIdx.y * 32;   // channel tile
    const float* src = in + (size_t)b * C_ * HW_;
    unsigned short* dst = out + (size_t)b * HW_ * C_;
    const int t = threadIdx.x;
    const int j = t & 31;
    const int i = t >> 5;

    #pragma unroll
    for (int p = 0; p < 4; ++p) {
        int cc = i + p * 8;
        float v = src[(size_t)(c0 + cc) * HW_ + (m0 + j)];   // coalesced in m
        tile[cc][j] = f32_to_bf16_rne(v);
    }
    __syncthreads();
    #pragma unroll
    for (int p = 0; p < 4; ++p) {
        int mm = i + p * 8;
        dst[(size_t)(m0 + mm) * C_ + (c0 + j)] = tile[j][mm]; // coalesced in c
    }
}

// ---------------------------------------------------------------------------
// Kernel 1: all-pairs correlation GEMM via WMMA bf16 with TDM staging.
//   corr0[b, p, q] = (1/16) * sum_c fmap1[b,c,p] * fmap2[b,c,q]
// Inputs pre-converted bf16 [B][HW][C].  Block 256 threads = 8 waves (4Mx2N);
// macro tile 128(M) x 64(N); wave tile 32x32 = 2x2 WMMA.  K staged in two
// 128-wide chunks: wave 0 issues TENSOR_LOAD_TO_LDS for the A (128x128) and
// B (64x128) tiles, waits TENSORcnt, then the workgroup barrier releases
// 16 v_wmma per chunk fed by ds_load_b128 fragments.
// ---------------------------------------------------------------------------
#define KC_ 128
__global__ __launch_bounds__(256) void corr_gemm_wmma(
    const unsigned short* __restrict__ f1T,
    const unsigned short* __restrict__ f2T,
    float* __restrict__ corr0) {
    __shared__ unsigned short sA[128 * KC_];   // [m][k], 256B rows (32 KB)
    __shared__ unsigned short sB[64 * KC_];    // [n][k], 256B rows (16 KB)

    const int b  = blockIdx.z;
    const int m0 = blockIdx.y * 128;
    const int n0 = blockIdx.x * 64;
    const int t    = threadIdx.x;
    const int lane = t & 31;
    const int wave = t >> 5;
    const int waveM = wave & 3;   // 0..3
    const int waveN = wave >> 2;  // 0..1
    const int lrow  = lane & 15;
    const int hi    = lane >> 4;

    const unsigned short* A  = f1T + (size_t)b * HW_ * C_;
    const unsigned short* Bm = f2T + (size_t)b * HW_ * C_;

    v8f acc[2][2];
    {
        v8f z = {};
        acc[0][0] = z; acc[0][1] = z; acc[1][0] = z; acc[1][1] = z;
    }

    const unsigned ldsA = (unsigned)(uintptr_t)(void*)&sA[0];
    const unsigned ldsB = (unsigned)(uintptr_t)(void*)&sB[0];

    for (int k0 = 0; k0 < C_; k0 += KC_) {
        // ---- TDM staging: one wave issues both tile DMAs (EXEC ignored,
        //      wave-level op); TENSORcnt guards completion, then barrier.
        if (wave == 0) {
            // A tile: rows m0..m0+127 of [HW][C], columns k0..k0+127
            tdm_load_2d_bf16(ldsA, A + (size_t)(m0)*C_ + k0,
                             /*tensor_d0=*/C_, /*tensor_d1=*/HW_,
                             /*tile_d0=*/KC_, /*tile_d1=*/128,
                             /*stride_d0=*/C_);
            // B tile: rows n0..n0+63
            tdm_load_2d_bf16(ldsB, Bm + (size_t)(n0)*C_ + k0,
                             C_, HW_, KC_, 64, C_);
            __builtin_amdgcn_s_wait_tensorcnt(0);
        }
        __syncthreads();

        union Frag { v16bf v; v8us h[2]; };
        // 4 sub-chunks of K=32 per staged 128-wide chunk
        #pragma unroll
        for (int kk = 0; kk < 4; ++kk) {
            Frag fa[2], fb[2];
            // A fragment (16x32 bf16): lanes 0-15 hold K 0-7 + 16-23,
            // lanes 16-31 hold K 8-15 + 24-31 -> two ds_load_b128.
            #pragma unroll
            for (int mi = 0; mi < 2; ++mi) {
                int m = waveM * 32 + mi * 16 + lrow;
                const v8us* p = (const v8us*)(sA + m * KC_ + kk * 32 + hi * 8);
                fa[mi].h[0] = p[0];
                fa[mi].h[1] = p[2];
            }
            // B fragment (32x16 bf16): lanes 0-15 K 0-7 + 16-23, lanes 16-31 K 8-15 + 24-31.
            #pragma unroll
            for (int ni = 0; ni < 2; ++ni) {
                int n = waveN * 32 + ni * 16 + lrow;
                const v8us* p = (const v8us*)(sB + n * KC_ + kk * 32 + hi * 8);
                fb[ni].h[0] = p[0];
                fb[ni].h[1] = p[2];
            }
            #pragma unroll
            for (int mi = 0; mi < 2; ++mi)
                #pragma unroll
                for (int ni = 0; ni < 2; ++ni)
                    acc[mi][ni] = __builtin_amdgcn_wmma_f32_16x16x32_bf16(
                        false, fa[mi].v, false, fb[ni].v,
                        (short)0, acc[mi][ni], false, false);
        }
        __syncthreads();
    }

    // C/D layout: VGPR r holds M = r + (lane>=16 ? 8 : 0), N = lane%16.
    const float scale = 0.0625f;   // 1/sqrt(256)
    #pragma unroll
    for (int mi = 0; mi < 2; ++mi) {
        #pragma unroll
        for (int ni = 0; ni < 2; ++ni) {
            int q = n0 + waveN * 32 + ni * 16 + lrow;
            #pragma unroll
            for (int r = 0; r < 8; ++r) {
                int p = m0 + waveM * 32 + mi * 16 + r + hi * 8;
                corr0[((size_t)b * HW_ + p) * HW_ + q] = acc[mi][ni][r] * scale;
            }
        }
    }
}

// ---------------------------------------------------------------------------
// Kernel 2: 2x2 average pool over the target (h,w) dims of one pyramid level.
// ---------------------------------------------------------------------------
__global__ void pool2x2(const float* __restrict__ in, float* __restrict__ out,
                        int Hin, int Win, int total) {
    int tid = blockIdx.x * blockDim.x + threadIdx.x;
    if (tid >= total) return;
    int Wout = Win >> 1;
    int Hout = Hin >> 1;
    int x = tid % Wout;
    int y = (tid / Wout) % Hout;
    int n = tid / (Wout * Hout);
    const float* p = in + (size_t)n * Hin * Win + (size_t)(2 * y) * Win + 2 * x;
    out[tid] = 0.25f * (p[0] + p[1] + p[Win] + p[Win + 1]);
}

// ---------------------------------------------------------------------------
// Kernel 3: 81-point bilinear lookup over 4 levels (zero padding outside).
// ---------------------------------------------------------------------------
__device__ __forceinline__ float corr_tap(const float* cf, int Hl, int Wl,
                                          int xi, int yi) {
    if (xi < 0 || xi >= Wl || yi < 0 || yi >= Hl) return 0.0f;
    return cf[yi * Wl + xi];
}

__global__ void corr_lookup(const float* __restrict__ coords,
                            const float* __restrict__ l0,
                            const float* __restrict__ l1,
                            const float* __restrict__ l2,
                            const float* __restrict__ l3,
                            float* __restrict__ out) {
    int tid = blockIdx.x * blockDim.x + threadIdx.x;
    if (tid >= NQ_ * 81) return;
    int q    = tid % HW_;          // h*W8 + w (pixel-contiguous: coalesced)
    int rest = tid / HW_;
    int j    = rest % 81;
    int b    = rest / 81;

    float cx = coords[((size_t)b * 2 + 0) * HW_ + q];
    float cy = coords[((size_t)b * 2 + 1) * HW_ + q];
    float xo = (float)(j / 9) - 4.0f;   // meshgrid quirk: x varies along axis 0
    float yo = (float)(j % 9) - 4.0f;
    int n = b * HW_ + q;

    const float* lvls[4] = {l0, l1, l2, l3};
    float inv = 1.0f;
    #pragma unroll
    for (int i = 0; i < 4; ++i) {
        int Hl = H8_ >> i;
        int Wl = W8_ >> i;
        const float* cf = lvls[i] + (size_t)n * Hl * Wl;
        float px = cx * inv + xo;
        float py = cy * inv + yo;
        float x0f = floorf(px), y0f = floorf(py);
        float wx = px - x0f, wy = py - y0f;
        int x0 = (int)x0f, y0 = (int)y0f;
        float v00 = corr_tap(cf, Hl, Wl, x0,     y0);
        float v10 = corr_tap(cf, Hl, Wl, x0 + 1, y0);
        float v01 = corr_tap(cf, Hl, Wl, x0,     y0 + 1);
        float v11 = corr_tap(cf, Hl, Wl, x0 + 1, y0 + 1);
        float r = (1.0f - wy) * ((1.0f - wx) * v00 + wx * v10)
                +          wy * ((1.0f - wx) * v01 + wx * v11);
        out[((size_t)b * 324 + i * 81 + j) * HW_ + q] = r;
        inv *= 0.5f;
    }
}

// ---------------------------------------------------------------------------
// Kernel 4: convex upsample. Thread per (b, h, di, w, dj) -> writes c=0,1.
// ---------------------------------------------------------------------------
__global__ void convex_upsample(const float* __restrict__ flow,
                                const float* __restrict__ mask,
                                float* __restrict__ out) {
    int tid = blockIdx.x * blockDim.x + threadIdx.x;
    const int total = B_ * 64 * HW_;
    if (tid >= total) return;
    int dj = tid & 7;
    int w  = (tid >> 3) % W8_;
    int di = (tid / (8 * W8_)) & 7;
    int h  = (tid / (64 * W8_)) % H8_;
    int b  = tid / (64 * HW_);

    float mv[9];
    float mx = -INFINITY;
    #pragma unroll
    for (int k = 0; k < 9; ++k) {
        mv[k] = mask[((size_t)b * 576 + k * 64 + di * 8 + dj) * HW_ + h * W8_ + w];
        mx = fmaxf(mx, mv[k]);
    }
    float s = 0.0f;
    #pragma unroll
    for (int k = 0; k < 9; ++k) { mv[k] = __expf(mv[k] - mx); s += mv[k]; }
    float invs = 1.0f / s;

    #pragma unroll
    for (int c = 0; c < 2; ++c) {
        float acc = 0.0f;
        #pragma unroll
        for (int k = 0; k < 9; ++k) {
            int hh = h + k / 3 - 1;
            int ww = w + k % 3 - 1;
            float fv = 0.0f;
            if (hh >= 0 && hh < H8_ && ww >= 0 && ww < W8_)
                fv = 8.0f * flow[((size_t)b * 2 + c) * HW_ + hh * W8_ + ww];
            acc += mv[k] * fv;
        }
        out[(((size_t)b * 2 + c) * (8 * H8_) + 8 * h + di) * (size_t)(8 * W8_)
            + 8 * w + dj] = acc * invs;
    }
}

// ---------------------------------------------------------------------------
extern "C" void kernel_launch(void* const* d_in, const int* in_sizes, int n_in,
                              void* d_out, int out_size, void* d_ws, size_t ws_size,
                              hipStream_t stream) {
    (void)in_sizes; (void)n_in; (void)out_size; (void)ws_size;
    const float* fmap1  = (const float*)d_in[0];
    const float* fmap2  = (const float*)d_in[1];
    const float* coords = (const float*)d_in[2];
    const float* flow   = (const float*)d_in[3];
    const float* mask   = (const float*)d_in[4];
    float* out = (float*)d_out;

    // Workspace: pyramid levels back-to-back (f32), then bf16 fmaps
    float* l0 = (float*)d_ws;                       // [NQ, 56, 96]
    size_t s0 = (size_t)NQ_ * HW_;
    float* l1 = l0 + s0;                            // [NQ, 28, 48]
    size_t s1 = (size_t)NQ_ * (HW_ / 4);
    float* l2 = l1 + s1;                            // [NQ, 14, 24]
    size_t s2 = (size_t)NQ_ * (HW_ / 16);
    float* l3 = l2 + s2;                            // [NQ, 7, 12]
    size_t s3 = (size_t)NQ_ * (HW_ / 64);
    unsigned short* f1T = (unsigned short*)(l3 + s3);        // [B][HW][C] bf16
    unsigned short* f2T = f1T + (size_t)B_ * HW_ * C_;       // [B][HW][C] bf16

    // 0) f32 -> bf16 transpose-convert (one-time, ~11 MB)
    {
        dim3 gc(HW_ / 32, C_ / 32, B_);
        convert_transpose_bf16<<<gc, 256, 0, stream>>>(fmap1, f1T);
        convert_transpose_bf16<<<gc, 256, 0, stream>>>(fmap2, f2T);
    }

    // 1) correlation GEMM (WMMA + TDM staging)
    dim3 ggemm(HW_ / 64, HW_ / 128, B_);
    corr_gemm_wmma<<<ggemm, 256, 0, stream>>>(f1T, f2T, l0);

    // 2) pyramid pooling
    {
        int total1 = NQ_ * (HW_ / 4);
        pool2x2<<<(total1 + 255) / 256, 256, 0, stream>>>(l0, l1, H8_, W8_, total1);
        int total2 = NQ_ * (HW_ / 16);
        pool2x2<<<(total2 + 255) / 256, 256, 0, stream>>>(l1, l2, H8_ / 2, W8_ / 2, total2);
        int total3 = NQ_ * (HW_ / 64);
        pool2x2<<<(total3 + 255) / 256, 256, 0, stream>>>(l2, l3, H8_ / 4, W8_ / 4, total3);
    }

    // 3) lookups -> first output chunk [B, 324, H8, W8]
    {
        int total = NQ_ * 81;
        corr_lookup<<<(total + 255) / 256, 256, 0, stream>>>(coords, l0, l1, l2, l3, out);
    }

    // 4) convex upsample -> second output chunk [B, 2, 8*H8, 8*W8]
    {
        float* out2 = out + (size_t)B_ * 324 * HW_;
        int total = B_ * 64 * HW_;
        convex_upsample<<<(total + 255) / 256, 256, 0, stream>>>(flow, mask, out2);
    }
}
